// KNNModel_63763084477134
// MI455X (gfx1250) — matrix operand order repaired and checked by reading, not compiled, not run on hardware
//
#include <hip/hip_runtime.h>
#include <hip/hip_bf16.h>

// nnan/ninf fast flags: min/max lower to bare v_min_num/v_max_num.
#pragma float_control(precise, off)

typedef __attribute__((ext_vector_type(16))) __bf16 v16bf;
typedef __attribute__((ext_vector_type(8)))  __bf16 v8bf;
typedef __attribute__((ext_vector_type(8)))  float  v8f;

#define B_      4
#define D_      128
#define H_      32
#define W_      32
#define Q_      4096          /* B*H*W                              */
#define N_      50000
#define NSPLIT_ 8             /* N chunks (one per blockIdx.y)      */
#define TILES_  392           /* tiles per chunk (even -> ping-pong)*/
#define CHUNK_  6272          /* TILES_ * 16                        */
#define NPAD_   50176         /* NSPLIT_ * CHUNK_                   */
#define NALLOC_ 50192         /* + one lookahead tile               */
#define OH_     512
#define OW_     512
#define KVAL_   5
#define BIGD_   1e30f

// Branchless insert of x into ascending 5-smallest list (values only; the
// reference discards kNN indices). 9 v_min/v_max ops.
__device__ __forceinline__ void insert5(float (&t)[KVAL_], float x) {
  float m;
  m = fminf(t[0], x); x = fmaxf(t[0], x); t[0] = m;
  m = fminf(t[1], x); x = fmaxf(t[1], x); t[1] = m;
  m = fminf(t[2], x); x = fmaxf(t[2], x); t[2] = m;
  m = fminf(t[3], x); x = fmaxf(t[3], x); t[3] = m;
  t[4] = fminf(t[4], x);
}

// ---- prep: bank -> bf16 (padded) + fp32 squared norms ----------------------
__global__ __launch_bounds__(128) void bank_prep_kernel(
    const float* __restrict__ bank, __bf16* __restrict__ bbf,
    float* __restrict__ b2) {
  __shared__ float red[128];
  const int n = blockIdx.x, d = threadIdx.x;
  float v = (n < N_) ? bank[(size_t)n * D_ + d] : 0.0f;
  bbf[(size_t)n * D_ + d] = (__bf16)v;
  red[d] = v * v;
  __syncthreads();
#pragma unroll
  for (int s = 64; s > 0; s >>= 1) {
    if (d < s) red[d] += red[d + s];
    __syncthreads();
  }
  if (d == 0) b2[n] = (n < N_) ? red[0] : BIGD_;  // pad rows never rank
}

// ---- prep: embeddings [B,D,h,w] -> queries [Q,D] bf16 + norms --------------
__global__ __launch_bounds__(128) void q_prep_kernel(
    const float* __restrict__ emb, __bf16* __restrict__ qbf,
    float* __restrict__ q2) {
  __shared__ float red[128];
  const int q = blockIdx.x, d = threadIdx.x;
  const int b = q >> 10, y = (q >> 5) & 31, x = q & 31;
  float v = emb[(((size_t)b * D_ + d) * H_ + y) * W_ + x];
  qbf[(size_t)q * D_ + d] = (__bf16)v;
  red[d] = v * v;
  __syncthreads();
#pragma unroll
  for (int s = 64; s > 0; s >>= 1) {
    if (d < s) red[d] += red[d + s];
    __syncthreads();
  }
  if (d == 0) q2[q] = red[0];
}

// One 16x16 distance tile: preload next B tile (double buffer), 4 chained
// bf16 WMMAs over K=128, epilogue d^2 + threshold-gated top-5 insert.
__device__ __forceinline__ void knn_tile(
    const v16bf (&a)[4], const v16bf (&cur)[4], float b2n,
    v16bf (&nxt)[4], float& nb2, const __bf16* nxt_ptr,
    const float* nb2_ptr, const float (&q2v)[8], float (&t)[8][KVAL_],
    float& thresh) {
  // issue next-tile loads first (software pipeline)
#pragma unroll
  for (int kk = 0; kk < 4; ++kk) nxt[kk] = *(const v16bf*)(nxt_ptr + kk * 32);
  nb2 = *nb2_ptr;
  __builtin_prefetch(nxt_ptr + 16 * D_, 0, 1);  // two tiles ahead
  v8f acc = {0.f, 0.f, 0.f, 0.f, 0.f, 0.f, 0.f, 0.f};
#pragma unroll
  for (int kk = 0; kk < 4; ++kk)
    acc = __builtin_amdgcn_wmma_f32_16x16x32_bf16(
        false, a[kk], false, cur[kk], (short)0, acc, false, false);
  float d2[8];
#pragma unroll
  for (int v = 0; v < 8; ++v)
    d2[v] = fmaf(acc[v], -2.0f, q2v[v] + b2n);  // ||q||^2+||b||^2-2q.b
  // early-out: can any of the 8 candidates enter any of this lane's lists?
  float m01 = fminf(d2[0], d2[1]), m23 = fminf(d2[2], d2[3]);
  float m45 = fminf(d2[4], d2[5]), m67 = fminf(d2[6], d2[7]);
  float m8 = fminf(fminf(m01, m23), fminf(m45, m67));
  if (m8 < thresh) {  // rare after warm-up
#pragma unroll
    for (int v = 0; v < 8; ++v) insert5(t[v], d2[v]);
    float x01 = fmaxf(t[0][4], t[1][4]), x23 = fmaxf(t[2][4], t[3][4]);
    float x45 = fmaxf(t[4][4], t[5][4]), x67 = fmaxf(t[6][4], t[7][4]);
    thresh = fmaxf(fmaxf(x01, x23), fmaxf(x45, x67));
  }
}

// ---- fused GEMM (bf16 WMMA) + per-query partial 5-NN -----------------------
// grid = (Q_/128 query-groups, NSPLIT_ N-chunks); block = 8 wave32.
// All 8 waves stream the SAME bank chunk (WGP$ serves 7/8 of B reads -> 8x
// less L2 traffic) while each wave owns its own 16-query tile.
__global__ __launch_bounds__(256) void knn_topk_kernel(
    const __bf16* __restrict__ qbf, const float* __restrict__ q2,
    const __bf16* __restrict__ bbf, const float* __restrict__ b2,
    float* __restrict__ partials) {
  __shared__ float cand[8 * 16 * 16 * KVAL_];  // [wave][row][col][5]

  const int tid  = threadIdx.x;
  const int wave = tid >> 5;
  const int lane = tid & 31;
  const int hi   = lane >> 4;   // half-wave select
  const int cl   = lane & 15;   // column (B) or row (A) within half
  const int qbase = blockIdx.x * 128 + wave * 16;  // per-wave query tile
  const int split = blockIdx.y;

  // A tile 16x128 bf16, contiguous K: lanes0-15 K{0-15}, lanes16-31 K{16-31}
  v16bf a[4];
  {
    const __bf16* qrow = qbf + (size_t)(qbase + cl) * D_ + (hi ? 16 : 0);
#pragma unroll
    for (int kk = 0; kk < 4; ++kk) {
      a[kk] = *(const v16bf*)(qrow + kk * 32);
    }
  }
  // q2 for the 8 accumulator rows this lane holds (M = v + 8*hi)
  float q2v[8];
#pragma unroll
  for (int v = 0; v < 8; ++v) q2v[v] = q2[qbase + v + 8 * hi];

  float t[8][KVAL_];
#pragma unroll
  for (int v = 0; v < 8; ++v)
#pragma unroll
    for (int j = 0; j < KVAL_; ++j) t[v][j] = BIGD_;
  float thresh = BIGD_;

  const int nStart = split * CHUNK_;  // shared by all 8 waves of this block
  // B tile: lane holds column n = nbase+cl, contiguous K half (hi ? 16..31 : 0..15)
  const __bf16* bptr = bbf + (size_t)(nStart + cl) * D_ + (hi ? 16 : 0);
  const float* b2p = b2 + nStart + cl;

  v16bf bt0[4], bt1[4];
  float b20, b21;
#pragma unroll
  for (int kk = 0; kk < 4; ++kk) bt0[kk] = *(const v16bf*)(bptr + kk * 32);
  b20 = b2p[0];

  for (int tIdx = 0; tIdx < TILES_; tIdx += 2) {  // ping-pong double buffer
    knn_tile(a, bt0, b20, bt1, b21, bptr + 16 * D_, b2p + 16, q2v, t, thresh);
    knn_tile(a, bt1, b21, bt0, b20, bptr + 32 * D_, b2p + 32, q2v, t, thresh);
    bptr += 32 * D_;
    b2p  += 32;
    if ((tIdx & 31) == 30) __syncthreads();  // keep waves converged for WGP$
  }

  // spill per-(row,col-lane) top5 to LDS (per-wave regions)
#pragma unroll
  for (int v = 0; v < 8; ++v) {
    const int row = v + 8 * hi;
    float* dst = &cand[(((wave * 16 + row) * 16) + cl) * KVAL_];
#pragma unroll
    for (int j = 0; j < KVAL_; ++j) dst[j] = t[v][j];
  }
  __syncthreads();

  // reduce 16 column-lanes -> partial top5 per (wave, row); 128 threads
  if (tid < 128) {
    const int w = tid >> 4, r = tid & 15;
    float u[KVAL_] = {BIGD_, BIGD_, BIGD_, BIGD_, BIGD_};
    const float* src = &cand[((w * 16 + r) * 16) * KVAL_];
    for (int c = 0; c < 16 * KVAL_; ++c) insert5(u, src[c]);
    const int q = blockIdx.x * 128 + w * 16 + r;
    float* dst = &partials[((size_t)split * Q_ + q) * KVAL_];
#pragma unroll
    for (int j = 0; j < KVAL_; ++j) dst[j] = u[j];
  }
}

// ---- merge NSPLIT_ partial top5s -> final 5-NN mean distance ---------------
__global__ __launch_bounds__(256) void knn_merge_kernel(
    const float* __restrict__ partials, float* __restrict__ scores) {
  const int q = blockIdx.x * 256 + threadIdx.x;  // Q_
  float u[KVAL_] = {BIGD_, BIGD_, BIGD_, BIGD_, BIGD_};
#pragma unroll
  for (int s = 0; s < NSPLIT_; ++s) {
    const float* src = &partials[((size_t)s * Q_ + q) * KVAL_];
#pragma unroll
    for (int j = 0; j < KVAL_; ++j) insert5(u, src[j]);
  }
  float s = 0.0f;
#pragma unroll
  for (int j = 0; j < KVAL_; ++j) s += sqrtf(fmaxf(u[j], 1e-12f));
  scores[q] = s * (1.0f / KVAL_);
}

// ---- half-pixel bilinear 32x32 -> 512x512 ----------------------------------
__global__ __launch_bounds__(256) void resize_kernel(
    const float* __restrict__ s, float* __restrict__ out) {
  const int idx = blockIdx.x * 256 + threadIdx.x;  // B*OH*OW
  const int b  = idx >> 18;
  const int oy = (idx >> 9) & (OH_ - 1);
  const int ox = idx & (OW_ - 1);
  const float scale = (float)H_ / (float)OH_;
  float sy = (oy + 0.5f) * scale - 0.5f;
  float sx = (ox + 0.5f) * scale - 0.5f;
  float fy0 = floorf(sy), fx0 = floorf(sx);
  int y0 = (int)fy0, x0 = (int)fx0;
  float wy = sy - fy0, wx = sx - fx0;
  int y0c = min(max(y0, 0), H_ - 1), y1c = min(max(y0 + 1, 0), H_ - 1);
  int x0c = min(max(x0, 0), W_ - 1), x1c = min(max(x0 + 1, 0), W_ - 1);
  const float* sb = s + b * (H_ * W_);
  float v00 = sb[y0c * W_ + x0c], v01 = sb[y0c * W_ + x1c];
  float v10 = sb[y1c * W_ + x0c], v11 = sb[y1c * W_ + x1c];
  out[idx] = (1.f - wy) * ((1.f - wx) * v00 + wx * v01) +
             wy * ((1.f - wx) * v10 + wx * v11);
}

extern "C" void kernel_launch(void* const* d_in, const int* in_sizes, int n_in,
                              void* d_out, int out_size, void* d_ws,
                              size_t ws_size, hipStream_t stream) {
  const float* emb  = (const float*)d_in[0];  // [4,128,32,32]
  const float* bank = (const float*)d_in[1];  // [50000,128]
  float* out = (float*)d_out;                 // [4,1,512,512]

  // workspace carve-out (~14.9 MB), 256B aligned
  char* ws = (char*)d_ws;
  size_t off = 0;
  auto carve = [&](size_t bytes) {
    char* p = ws + off;
    off += (bytes + 255) & ~(size_t)255;
    return p;
  };
  __bf16* bankbf  = (__bf16*)carve((size_t)NALLOC_ * D_ * sizeof(__bf16));
  __bf16* qbf     = (__bf16*)carve((size_t)Q_ * D_ * sizeof(__bf16));
  float*  b2      = (float*)carve((size_t)NALLOC_ * sizeof(float));
  float*  q2      = (float*)carve((size_t)Q_ * sizeof(float));
  float*  partial = (float*)carve((size_t)NSPLIT_ * Q_ * KVAL_ * sizeof(float));
  float*  sc      = (float*)carve((size_t)Q_ * sizeof(float));
  (void)in_sizes; (void)n_in; (void)out_size; (void)ws_size;

  hipLaunchKernelGGL(bank_prep_kernel, dim3(NALLOC_), dim3(D_), 0, stream,
                     bank, bankbf, b2);
  hipLaunchKernelGGL(q_prep_kernel, dim3(Q_), dim3(D_), 0, stream,
                     emb, qbf, q2);
  hipLaunchKernelGGL(knn_topk_kernel, dim3(Q_ / 128, NSPLIT_), dim3(256), 0,
                     stream, qbf, q2, bankbf, b2, partial);
  hipLaunchKernelGGL(knn_merge_kernel, dim3(Q_ / 256), dim3(256), 0, stream,
                     partial, sc);
  hipLaunchKernelGGL(resize_kernel, dim3((B_ * OH_ * OW_) / 256), dim3(256), 0,
                     stream, sc, out);
}
